// MultiHeadAttention_70866960384614
// MI455X (gfx1250) — matrix operand order repaired and checked by reading, not compiled, not run on hardware
//
#include <hip/hip_runtime.h>
#include <hip/hip_bf16.h>

// ---------------------------------------------------------------------------
// MHA forward on gfx1250 (CDNA5): bf16 WMMA GEMMs + flash attention.
// D = A(16x32 bf16) x B(32x16 bf16) + C(16x16 f32) via v_wmma_f32_16x16x32_bf16
// Softmax row reductions use v_permlane16_b32 (VALU) instead of ds_bpermute.
// ---------------------------------------------------------------------------

typedef __attribute__((ext_vector_type(16))) __bf16 bf16x16;
typedef __attribute__((ext_vector_type(8)))  float   v8f;

#define DM     1024
#define S_LEN  2048
#define NB     2
#define NH     16
#define DK     64
#define M_TOT  4096   // NB * S_LEN

union Frag {
    bf16x16 v;
    uint4   u[2];    // 32 bytes = 16 bf16 per lane
};

// ---------------------------------------------------------------------------
// xor-butterfly within each 16-lane group via v_permlane16_b32 (pure VALU).
// Lane-select nibbles encode  dst_lane i reads lane (i ^ off)  in its group.
// ---------------------------------------------------------------------------
__device__ __forceinline__ float permlane16_xor(float x, unsigned sel_lo,
                                                unsigned sel_hi) {
    int xi = __builtin_bit_cast(int, x);
    int r  = __builtin_amdgcn_permlane16(xi, xi, sel_lo, sel_hi, false, false);
    return __builtin_bit_cast(float, r);
}

__device__ __forceinline__ float rowmax16(float v) {
    v = fmaxf(v, permlane16_xor(v, 0x67452301u, 0xEFCDAB89u)); // xor 1
    v = fmaxf(v, permlane16_xor(v, 0x54761032u, 0xDCFE98BAu)); // xor 2
    v = fmaxf(v, permlane16_xor(v, 0x32107654u, 0xBA98FEDCu)); // xor 4
    v = fmaxf(v, permlane16_xor(v, 0xFEDCBA98u, 0x76543210u)); // xor 8
    return v;
}

__device__ __forceinline__ float rowsum16(float v) {
    v += permlane16_xor(v, 0x67452301u, 0xEFCDAB89u); // xor 1
    v += permlane16_xor(v, 0x54761032u, 0xDCFE98BAu); // xor 2
    v += permlane16_xor(v, 0x32107654u, 0xBA98FEDCu); // xor 4
    v += permlane16_xor(v, 0xFEDCBA98u, 0x76543210u); // xor 8
    return v;
}

// ---------------------------------------------------------------------------
// f32 -> bf16 conversion (grid-stride)
// ---------------------------------------------------------------------------
__global__ void f32_to_bf16_kernel(const float* __restrict__ in,
                                   __bf16* __restrict__ out, int n) {
    int i = blockIdx.x * blockDim.x + threadIdx.x;
    int stride = gridDim.x * blockDim.x;
    for (; i < n; i += stride) out[i] = (__bf16)in[i];
}

// ---------------------------------------------------------------------------
// GEMM: Y[m,n] = sum_k X[m,k] * W[n,k] + bias[n]   (X: [4096,1024] bf16,
// W: [1024,1024] bf16 row-major so W-rows are contiguous K => perfect B-frags)
// MODE 0: write bf16 head-split   [B,H,S,DK]
// MODE 1: write bf16 head-split transposed [B,H,DK,S]   (for V)
// MODE 2: write f32 row-major     [B,S,DM]              (final output)
// Block = 256 threads = 8 waves; block tile 128x128; wave tile 64x32.
// ---------------------------------------------------------------------------
template<int MODE>
__global__ __launch_bounds__(256) void gemm_wmma(
    const __bf16* __restrict__ X,
    const __bf16* __restrict__ W,
    const float*  __restrict__ bias,
    void* __restrict__ outp)
{
    const int lane = threadIdx.x & 31;
    const int wave = threadIdx.x >> 5;     // 0..7
    const int lhi  = lane >> 4;            // 0/1 (half-wave)
    const int lrw  = lane & 15;

    const int mBase = blockIdx.x * 128 + (wave >> 2) * 64;
    const int nBase = blockIdx.y * 128 + (wave & 3) * 32;

    v8f zero = {};
    v8f acc[4][2];
#pragma unroll
    for (int i = 0; i < 4; ++i)
#pragma unroll
        for (int j = 0; j < 2; ++j) acc[i][j] = zero;

    // A-frag lane base: lane holds row M=lrw; K elems {0-7,16-23} (lhi=0)
    // or {8-15,24-31} (lhi=1)
    const __bf16* aBase[4];
#pragma unroll
    for (int mi = 0; mi < 4; ++mi)
        aBase[mi] = X + (size_t)(mBase + mi * 16 + lrw) * DM + lhi * 8;
    // B-frag lane base: lane holds col N=lrw; K elems {0-7,16-23} / {8-15,24-31}
    const __bf16* bBase[2];
#pragma unroll
    for (int ni = 0; ni < 2; ++ni)
        bBase[ni] = W + (size_t)(nBase + ni * 16 + lrw) * DM + lhi * 8;

    for (int k0 = 0; k0 < DM; k0 += 32) {
        Frag a[4], b[2];
#pragma unroll
        for (int mi = 0; mi < 4; ++mi) {
            const uint4* p = (const uint4*)(aBase[mi] + k0);
            a[mi].u[0] = p[0];     // K+0..7
            a[mi].u[1] = p[2];     // K+16..23 (skip 16 elems = 2 x uint4)
        }
#pragma unroll
        for (int ni = 0; ni < 2; ++ni) {
            const uint4* p = (const uint4*)(bBase[ni] + k0);
            b[ni].u[0] = p[0];
            b[ni].u[1] = p[2];
        }
        if (k0 + 64 < DM) {        // warm L2/WGP$ for the next-but-one K step
            __builtin_prefetch(aBase[0] + k0 + 64, 0, 1);
            __builtin_prefetch(bBase[0] + k0 + 64, 0, 1);
        }
#pragma unroll
        for (int mi = 0; mi < 4; ++mi)
#pragma unroll
            for (int ni = 0; ni < 2; ++ni)
                acc[mi][ni] = __builtin_amdgcn_wmma_f32_16x16x32_bf16(
                    false, a[mi].v, false, b[ni].v,
                    (short)0, acc[mi][ni], false, false);
    }

    // epilogue: C layout lane=col N, VGPR r = row (r + 8*lhi)
#pragma unroll
    for (int ni = 0; ni < 2; ++ni) {
        const int   n  = nBase + ni * 16 + lrw;
        const float bn = bias[n];
#pragma unroll
        for (int mi = 0; mi < 4; ++mi) {
#pragma unroll
            for (int r = 0; r < 8; ++r) {
                const int   m   = mBase + mi * 16 + r + 8 * lhi;
                const float val = acc[mi][ni][r] + bn;
                if (MODE == 2) {
                    ((float*)outp)[(size_t)m * DM + n] = val;
                } else {
                    const int bb = m >> 11;          // / S_LEN
                    const int s  = m & (S_LEN - 1);
                    const int h  = n >> 6;           // / DK
                    const int d  = n & (DK - 1);
                    size_t idx;
                    if (MODE == 0)
                        idx = (((size_t)bb * NH + h) * S_LEN + s) * DK + d;
                    else
                        idx = (((size_t)bb * NH + h) * DK + d) * S_LEN + s;
                    ((__bf16*)outp)[idx] = (__bf16)val;
                }
            }
        }
    }
}

// ---------------------------------------------------------------------------
// Flash attention with |scores|: per wave one 16-query tile, keys in chunks
// of 32, online softmax, P staged via LDS (C-layout -> A-fragment transpose).
// Q/K: [BH, S, DK] bf16;  Vt: [BH, DK, S] bf16 (pre-transposed);
// ctx out: [B, S, DM] bf16.
// ---------------------------------------------------------------------------
__global__ __launch_bounds__(128) void flash_attn_kernel(
    const __bf16* __restrict__ Qp,
    const __bf16* __restrict__ Kp,
    const __bf16* __restrict__ Vt,
    __bf16* __restrict__ ctx)
{
    __shared__ __align__(16) __bf16 sP[4][16][32];   // per-wave P tile

    const int lane = threadIdx.x & 31;
    const int wave = threadIdx.x >> 5;    // 0..3
    const int lhi  = lane >> 4;
    const int lrw  = lane & 15;

    const int bh = blockIdx.y;            // 0..31
    const int bb = bh >> 4;
    const int h  = bh & 15;
    const int q0 = blockIdx.x * 64 + wave * 16;

    const __bf16* Q = Qp + (size_t)bh * S_LEN * DK;
    const __bf16* K = Kp + (size_t)bh * S_LEN * DK;
    const __bf16* V = Vt + (size_t)bh * DK * S_LEN;

    // Q A-fragments, loaded once (d_k = 64 -> two 16x32 K-steps)
    Frag qa[2];
#pragma unroll
    for (int ks = 0; ks < 2; ++ks) {
        const uint4* p =
            (const uint4*)(Q + (size_t)(q0 + lrw) * DK + ks * 32 + lhi * 8);
        qa[ks].u[0] = p[0];
        qa[ks].u[1] = p[2];
    }

    v8f zero = {};
    v8f o[4];
#pragma unroll
    for (int i = 0; i < 4; ++i) o[i] = zero;
    float mrow[8], lsum[8];
#pragma unroll
    for (int r = 0; r < 8; ++r) { mrow[r] = -1e30f; lsum[r] = 0.f; }

    const float scale = 0.125f;   // 1/sqrt(64)

    for (int t0 = 0; t0 < S_LEN; t0 += 32) {
        // ---- scores: two 16x16 tiles (keys t0..+15, t0+16..+31) ----
        v8f sc[2];
        sc[0] = zero; sc[1] = zero;
#pragma unroll
        for (int j = 0; j < 2; ++j) {
#pragma unroll
            for (int ks = 0; ks < 2; ++ks) {
                Frag bk;   // B[k][n] = K[t0+j*16+n][ks*32+k] -> contiguous
                const uint4* p = (const uint4*)(K +
                    (size_t)(t0 + j * 16 + lrw) * DK + ks * 32 + lhi * 8);
                bk.u[0] = p[0];
                bk.u[1] = p[2];
                sc[j] = __builtin_amdgcn_wmma_f32_16x16x32_bf16(
                    false, qa[ks].v, false, bk.v, (short)0, sc[j], false, false);
            }
        }
        // ---- |s|/8, chunk row-max (VALU permlane butterflies) ----
        float cmax[8];
#pragma unroll
        for (int r = 0; r < 8; ++r) {
            float a0 = fabsf(sc[0][r]) * scale;
            float a1 = fabsf(sc[1][r]) * scale;
            sc[0][r] = a0; sc[1][r] = a1;
            cmax[r] = rowmax16(fmaxf(a0, a1));
        }
        // ---- online softmax update ----
#pragma unroll
        for (int r = 0; r < 8; ++r) {
            const float mn    = fmaxf(mrow[r], cmax[r]);
            const float alpha = __expf(mrow[r] - mn);
            const float p0    = __expf(sc[0][r] - mn);
            const float p1    = __expf(sc[1][r] - mn);
            sc[0][r] = p0; sc[1][r] = p1;
            const float cs = rowsum16(p0 + p1);
            lsum[r] = lsum[r] * alpha + cs;
            mrow[r] = mn;
#pragma unroll
            for (int di = 0; di < 4; ++di) o[di][r] *= alpha;
        }
        // ---- stage P (16x32 bf16) through LDS to get A-fragment layout ----
#pragma unroll
        for (int r = 0; r < 8; ++r) {
            sP[wave][r + 8 * lhi][lrw]      = (__bf16)sc[0][r];
            sP[wave][r + 8 * lhi][16 + lrw] = (__bf16)sc[1][r];
        }
        __syncthreads();
        Frag pa;
        {
            const uint4* p = (const uint4*)(&sP[wave][lrw][lhi * 8]);
            pa.u[0] = p[0];
            pa.u[1] = p[2];
        }
        __syncthreads();
        // ---- o += P x V : B[k][n] = Vt[di*16+n][t0+k] -> contiguous ----
#pragma unroll
        for (int di = 0; di < 4; ++di) {
            Frag bv;
            const uint4* p = (const uint4*)(V +
                (size_t)(di * 16 + lrw) * S_LEN + t0 + lhi * 8);
            bv.u[0] = p[0];
            bv.u[1] = p[2];
            o[di] = __builtin_amdgcn_wmma_f32_16x16x32_bf16(
                false, pa.v, false, bv.v, (short)0, o[di], false, false);
        }
    }

    // ---- normalize and write ctx[bb][s][h*64 + d] ----
#pragma unroll
    for (int di = 0; di < 4; ++di) {
        const int d = h * 64 + di * 16 + lrw;
#pragma unroll
        for (int r = 0; r < 8; ++r) {
            const int s = q0 + r + 8 * lhi;
            ctx[((size_t)bb * S_LEN + s) * DM + d] =
                (__bf16)(o[di][r] / lsum[r]);
        }
    }
}

// ---------------------------------------------------------------------------
// launch
// ---------------------------------------------------------------------------
extern "C" void kernel_launch(void* const* d_in, const int* in_sizes, int n_in,
                              void* d_out, int out_size, void* d_ws, size_t ws_size,
                              hipStream_t stream) {
    const float* q  = (const float*)d_in[0];
    const float* k  = (const float*)d_in[1];
    const float* v  = (const float*)d_in[2];
    const float* Wq = (const float*)d_in[3];
    const float* bq = (const float*)d_in[4];
    const float* Wk = (const float*)d_in[5];
    const float* bk = (const float*)d_in[6];
    const float* Wv = (const float*)d_in[7];
    const float* bv = (const float*)d_in[8];
    const float* Wo = (const float*)d_in[9];
    const float* bo = (const float*)d_in[10];

    __bf16* ws = (__bf16*)d_ws;
    const size_t XE = (size_t)M_TOT * DM;   // activation elems
    const size_t WE = (size_t)DM * DM;      // weight elems
    __bf16* qb  = ws;
    __bf16* kb  = qb  + XE;
    __bf16* vb  = kb  + XE;
    __bf16* wqb = vb  + XE;
    __bf16* wkb = wqb + WE;
    __bf16* wvb = wkb + WE;
    __bf16* wob = wvb + WE;
    __bf16* Qp  = wob + WE;
    __bf16* Kp  = Qp  + XE;
    __bf16* Vt  = Kp  + XE;
    __bf16* ctx = Vt  + XE;
    // total: 64 MiB of workspace

    f32_to_bf16_kernel<<<1024, 256, 0, stream>>>(q,  qb,  (int)XE);
    f32_to_bf16_kernel<<<1024, 256, 0, stream>>>(k,  kb,  (int)XE);
    f32_to_bf16_kernel<<<1024, 256, 0, stream>>>(v,  vb,  (int)XE);
    f32_to_bf16_kernel<<<1024, 256, 0, stream>>>(Wq, wqb, (int)WE);
    f32_to_bf16_kernel<<<1024, 256, 0, stream>>>(Wk, wkb, (int)WE);
    f32_to_bf16_kernel<<<1024, 256, 0, stream>>>(Wv, wvb, (int)WE);
    f32_to_bf16_kernel<<<1024, 256, 0, stream>>>(Wo, wob, (int)WE);

    dim3 gg(M_TOT / 128, DM / 128);
    gemm_wmma<0><<<gg, 256, 0, stream>>>(qb, wqb, bq, Qp);
    gemm_wmma<0><<<gg, 256, 0, stream>>>(kb, wkb, bk, Kp);
    gemm_wmma<1><<<gg, 256, 0, stream>>>(vb, wvb, bv, Vt);

    dim3 ga(S_LEN / 64, NB * NH);
    flash_attn_kernel<<<ga, 128, 0, stream>>>(Qp, Kp, Vt, ctx);

    gemm_wmma<2><<<gg, 256, 0, stream>>>(ctx, wob, bo, d_out);
}
